// MPNNTransformIdentity_54039278518619
// MI455X (gfx1250) — compile-verified, matrix operand order checked by reading, not run on hardware
//
#include <hip/hip_runtime.h>
#include <hip/hip_bf16.h>

typedef _Float16 half_t;
typedef _Float16 v16h __attribute__((ext_vector_type(16)));
typedef _Float16 v8h  __attribute__((ext_vector_type(8)));
typedef float    v8f  __attribute__((ext_vector_type(8)));

#define BATCH  1024
#define NVERT  256
#define HID    128
#define F1     8
#define GINUM  264           // [h(128) | message(128) | jets(8)]
#define NITER  3
#define KPAD   160           // 128 h + 8 jets + 24 zero = 5 chunks of K32
#define KC_BIG 5
#define KC_H   4
#define ROWS   (BATCH * NVERT)   // 262144
#define MTILE  32                // rows per block for big kernels
#define GBLK   (ROWS / MTILE)    // 8192 blocks
#define BIGSZ  (8 * KC_BIG * 32 * 16)  // packed halves per big weight
#define MIDSZ  (8 * KC_H   * 32 * 16)  // packed halves per 128x128 weight

// ---------------------------------------------------------------- helpers ---

__device__ __forceinline__ v8f wmma_f16(v16h a, v16h b, v8f c) {
  // D = A(16x32 f16) x B(32x16 f16) + C(16x16 f32)
  return __builtin_amdgcn_wmma_f32_16x16x32_f16(
      /*neg_a=*/false, a, /*neg_b=*/false, b,
      /*c_mod=*/(short)0, c, /*reuse_a=*/false, /*reuse_b=*/false);
}

// A-fragment (16x32 f16) from an LDS tile [rows x pitch halves].
// ISA 7.12.2: lanes 0-15 hold K {0..7,16..23}, lanes 16-31 hold K {8..15,24..31};
// row m = lane & 15. Each half of the fragment is a contiguous 8-half run.
__device__ __forceinline__ v16h load_afrag(const half_t* lds, int lane, int kc, int pitch) {
  const int m   = lane & 15;
  const int off = (lane & 16) ? 8 : 0;
  const half_t* base = lds + m * pitch + kc * 32;
  v8h lo = *(const v8h*)(base + off);
  v8h hi = *(const v8h*)(base + 16 + off);
  v16h a;
#pragma unroll
  for (int j = 0; j < 8; ++j) { a[j] = lo[j]; a[j + 8] = hi[j]; }
  return a;
}

// B-fragment: pre-packed contiguous 16 halves per lane.
__device__ __forceinline__ v16h load_bfrag(const half_t* pk, int nt, int KC, int kc, int lane) {
  return *(const v16h*)(pk + ((size_t)((nt * KC + kc) * 32 + lane) << 4));
}

// Branchless activations: v_exp_f32 + v_rcp_f32, saturate correctly at +/-inf.
__device__ __forceinline__ float sigmoid_f(float x) {
  return __builtin_amdgcn_rcpf(1.0f + __builtin_amdgcn_exp2f(-1.44269504f * x));
}
__device__ __forceinline__ float tanh_f(float x) {
  return 1.0f - 2.0f * __builtin_amdgcn_rcpf(1.0f + __builtin_amdgcn_exp2f(2.88539008f * x));
}

// ------------------------------------------------------------ weight pack ---
// mode 0: plain [128,128]; mode 1: rows 0..127 = W[0:128], 128..135 = W[256:264],
// rest zero (h|jets padded); mode 2: rows = W[128+k] (message middle block).
__global__ void pack_kernel(half_t* __restrict__ dst, const float* __restrict__ src,
                            int KC, int mode) {
  int p     = blockIdx.x * blockDim.x + threadIdx.x;
  int j     = p & 15;
  int lane  = (p >> 4) & 31;
  int chunk = p >> 9;
  int kc    = chunk % KC;
  int nt    = chunk / KC;
  int kb    = (j < 8) ? j : (j + 8);
  if (lane & 16) kb += 8;
  int k = kc * 32 + kb;
  int n = (nt << 4) + (lane & 15);
  float v = 0.0f;
  if (mode == 0)       v = src[k * HID + n];
  else if (mode == 1) { if (k < 128) v = src[k * HID + n];
                        else if (k < 128 + F1) v = src[(256 + (k - 128)) * HID + n]; }
  else                 v = src[(128 + k) * HID + n];
  dst[p] = (half_t)v;
}

// ----------------------------------------------------------------- embed ----
// h = tanh(jets @ W_emb + b_emb); also per-block (32-row) column partials.
__global__ __launch_bounds__(256) void embed_kernel(
    const float* __restrict__ jets, const float* __restrict__ W_emb,
    const float* __restrict__ b_emb, half_t* __restrict__ hout,
    float* __restrict__ hpart) {
  __shared__ float jls[MTILE * F1];   // 256 floats
  __shared__ float red[256];
  const int tid = threadIdx.x, bid = blockIdx.x;
  const int row0 = bid * MTILE;
  jls[tid] = jets[row0 * F1 + tid];
  __syncthreads();
  const int col = tid & 127;
  const int hb  = tid >> 7;           // 0/1 -> rows hb*16 .. +15
  float wv[F1];
#pragma unroll
  for (int f = 0; f < F1; ++f) wv[f] = W_emb[f * HID + col];
  const float bias = b_emb[col];
  float csum = 0.0f;
#pragma unroll
  for (int rr = 0; rr < 16; ++rr) {
    int m = hb * 16 + rr;
    float acc = bias;
#pragma unroll
    for (int f = 0; f < F1; ++f) acc += jls[m * F1 + f] * wv[f];
    float hv = tanh_f(acc);
    hout[(size_t)(row0 + m) * HID + col] = (half_t)hv;
    csum += hv;
  }
  red[tid] = csum;
  __syncthreads();
  if (tid < 128) hpart[bid * HID + col] = red[tid] + red[tid + 128];
}

// Deterministic 8-way partial reduction: hsum[b] = sum of 8 block partials.
__global__ void reduce8_kernel(float* __restrict__ dst, const float* __restrict__ src) {
  const int b = blockIdx.x, t = threadIdx.x;
  float s = 0.0f;
#pragma unroll
  for (int k = 0; k < 8; ++k) s += src[((b << 3) + k) * HID + t];
  dst[b * HID + t] = s;
}

// --------------------------------------------------------------- message ----
// msg = tanh(hsum@Wm + N*bm); c* = msg@W*_mid + b*  (per-batch constants)
__global__ __launch_bounds__(256) void message_kernel(
    const float* __restrict__ hsum, const half_t* __restrict__ WmP,
    const float* __restrict__ bm,
    const half_t* __restrict__ WzM, const half_t* __restrict__ WrM,
    const half_t* __restrict__ WhM,
    const float* __restrict__ bz, const float* __restrict__ br,
    const float* __restrict__ bh,
    float* __restrict__ cz, float* __restrict__ cr, float* __restrict__ ch) {
  __shared__ half_t xs[16 * HID];
  __shared__ half_t ms[16 * HID];
  const int tid = threadIdx.x, bid = blockIdx.x;
  const int brow0 = bid * 16;
  {  // vectorized f32->f16 staging: one 8-element chunk per thread
    int r = tid >> 4, c8 = (tid & 15) << 3;
    const float* src = &hsum[(size_t)(brow0 + r) * HID + c8];
    v8h d;
#pragma unroll
    for (int q = 0; q < 8; ++q) d[q] = (half_t)src[q];
    *(v8h*)&xs[r * HID + c8] = d;
  }
  __syncthreads();
  const int lane = tid & 31, w = tid >> 5;
  const int ncol = (w << 4) + (lane & 15);
  const int mbase = (lane & 16) ? 8 : 0;
  v8f am = {};
#pragma unroll
  for (int kc = 0; kc < KC_H; ++kc)
    am = wmma_f16(load_afrag(xs, lane, kc, HID), load_bfrag(WmP, w, KC_H, kc, lane), am);
  const float bmv = (float)NVERT * bm[ncol];
#pragma unroll
  for (int e = 0; e < 8; ++e)
    ms[(mbase + e) * HID + ncol] = (half_t)tanh_f(am[e] + bmv);
  __syncthreads();
  v8f az = {}, ar = {}, ah = {};
#pragma unroll
  for (int kc = 0; kc < KC_H; ++kc) {
    v16h a = load_afrag(ms, lane, kc, HID);
    az = wmma_f16(a, load_bfrag(WzM, w, KC_H, kc, lane), az);
    ar = wmma_f16(a, load_bfrag(WrM, w, KC_H, kc, lane), ar);
    ah = wmma_f16(a, load_bfrag(WhM, w, KC_H, kc, lane), ah);
  }
  const float bzv = bz[ncol], brv = br[ncol], bhv = bh[ncol];
#pragma unroll
  for (int e = 0; e < 8; ++e) {
    int b = brow0 + mbase + e;
    cz[b * HID + ncol] = az[e] + bzv;
    cr[b * HID + ncol] = ar[e] + brv;
    ch[b * HID + ncol] = ah[e] + bhv;
  }
}

// ------------------------------------------------------------------- GRU -----
// 32 rows per block: each wave keeps two A-tiles and two accumulators per gate
// so every B-fragment load feeds two WMMAs. All h traffic moves as b128.
__global__ __launch_bounds__(256) void gru_kernel(
    const half_t* __restrict__ hin, half_t* __restrict__ hout,
    const float* __restrict__ jets,
    const half_t* __restrict__ WzP, const half_t* __restrict__ WrP,
    const half_t* __restrict__ WhP,
    const float* __restrict__ cz, const float* __restrict__ cr,
    const float* __restrict__ ch,
    float* __restrict__ hpart) {
  __shared__ half_t xs[MTILE * KPAD];   // [h | jets | 0], later h_new
  __shared__ half_t rs[MTILE * KPAD];   // [r*h | jets | 0]
  const int tid = threadIdx.x, bid = blockIdx.x;
  const int row0 = bid * MTILE;
  const int batch = bid >> 3;           // 8 row-blocks per batch (N=256)
  // stage h: two b128 loads + two ds_store_b128 per thread
#pragma unroll
  for (int it = 0; it < 2; ++it) {
    int idx = tid + it * 256;           // 0..511 chunks of 8 halves
    int r = idx >> 4, c8 = (idx & 15) << 3;
    *(v8h*)&xs[r * KPAD + c8] = *(const v8h*)&hin[(size_t)(row0 + r) * HID + c8];
  }
#pragma unroll
  for (int it = 0; it < 4; ++it) {
    int idx = tid + it * 256, r = idx >> 5, c = idx & 31;
    half_t v = (c < F1) ? (half_t)jets[(row0 + r) * F1 + c] : (half_t)0.0f;
    xs[r * KPAD + 128 + c] = v;
    rs[r * KPAD + 128 + c] = v;
  }
  __syncthreads();

  const int lane = tid & 31, w = tid >> 5;
  const int ncol = (w << 4) + (lane & 15);
  const int mbase = (lane & 16) ? 8 : 0;

  // phase 1: z and r gates for both 16-row tiles
  v8f az0 = {}, az1 = {}, ar0 = {}, ar1 = {};
#pragma unroll
  for (int kc = 0; kc < KC_BIG; ++kc) {
    v16h a0 = load_afrag(xs, lane, kc, KPAD);
    v16h a1 = load_afrag(xs + 16 * KPAD, lane, kc, KPAD);
    v16h b  = load_bfrag(WzP, w, KC_BIG, kc, lane);
    az0 = wmma_f16(a0, b, az0);
    az1 = wmma_f16(a1, b, az1);
    b   = load_bfrag(WrP, w, KC_BIG, kc, lane);
    ar0 = wmma_f16(a0, b, ar0);
    ar1 = wmma_f16(a1, b, ar1);
  }
  const float czv = cz[batch * HID + ncol];
  const float crv = cr[batch * HID + ncol];
  float zv0[8], zv1[8];
#pragma unroll
  for (int e = 0; e < 8; ++e) {
    int m = mbase + e;
    zv0[e] = sigmoid_f(az0[e] + czv);
    float rv = sigmoid_f(ar0[e] + crv);
    rs[m * KPAD + ncol] = (half_t)(rv * (float)xs[m * KPAD + ncol]);
    zv1[e] = sigmoid_f(az1[e] + czv);
    rv = sigmoid_f(ar1[e] + crv);
    rs[(16 + m) * KPAD + ncol] = (half_t)(rv * (float)xs[(16 + m) * KPAD + ncol]);
  }
  __syncthreads();

  // phase 2: h_tilde + GRU blend. After the phase-1 barrier each wave touches
  // only its own 16-column strip of xs, so h_new can be written back into xs.
  v8f ah0 = {}, ah1 = {};
#pragma unroll
  for (int kc = 0; kc < KC_BIG; ++kc) {
    v16h a0 = load_afrag(rs, lane, kc, KPAD);
    v16h a1 = load_afrag(rs + 16 * KPAD, lane, kc, KPAD);
    v16h b  = load_bfrag(WhP, w, KC_BIG, kc, lane);
    ah0 = wmma_f16(a0, b, ah0);
    ah1 = wmma_f16(a1, b, ah1);
  }
  const float chv = ch[batch * HID + ncol];
  float colsum = 0.0f;
#pragma unroll
  for (int e = 0; e < 8; ++e) {
    int m = mbase + e;
    float ht   = tanh_f(ah0[e] + chv);
    float hold = (float)xs[m * KPAD + ncol];
    float hnew = (1.0f - zv0[e]) * hold + zv0[e] * ht;
    xs[m * KPAD + ncol] = (half_t)hnew;
    colsum += hnew;
    ht   = tanh_f(ah1[e] + chv);
    hold = (float)xs[(16 + m) * KPAD + ncol];
    hnew = (1.0f - zv1[e]) * hold + zv1[e] * ht;
    xs[(16 + m) * KPAD + ncol] = (half_t)hnew;
    colsum += hnew;
  }
  float other = __shfl_down(colsum, 16);
  if (lane < 16) hpart[bid * HID + ncol] = colsum + other;
  __syncthreads();
  // coalesced write-back: two ds_load_b128 + global_store_b128 per thread
#pragma unroll
  for (int it = 0; it < 2; ++it) {
    int idx = tid + it * 256;
    int r = idx >> 4, c8 = (idx & 15) << 3;
    *(v8h*)&hout[(size_t)(row0 + r) * HID + c8] = *(const v8h*)&xs[r * KPAD + c8];
  }
}

// ---------------------------------------------------------------- readout ---
__global__ __launch_bounds__(256) void readout1_kernel(
    const half_t* __restrict__ hin, const half_t* __restrict__ Wro1P,
    const float* __restrict__ bro1, float* __restrict__ hpart) {
  __shared__ half_t xs[MTILE * HID];
  const int tid = threadIdx.x, bid = blockIdx.x;
  const int row0 = bid * MTILE;
#pragma unroll
  for (int it = 0; it < 2; ++it) {
    int idx = tid + it * 256;
    int r = idx >> 4, c8 = (idx & 15) << 3;
    *(v8h*)&xs[r * HID + c8] = *(const v8h*)&hin[(size_t)(row0 + r) * HID + c8];
  }
  __syncthreads();
  const int lane = tid & 31, w = tid >> 5;
  const int ncol = (w << 4) + (lane & 15);
  v8f acc0 = {}, acc1 = {};
#pragma unroll
  for (int kc = 0; kc < KC_H; ++kc) {
    v16h b = load_bfrag(Wro1P, w, KC_H, kc, lane);
    acc0 = wmma_f16(load_afrag(xs, lane, kc, HID), b, acc0);
    acc1 = wmma_f16(load_afrag(xs + 16 * HID, lane, kc, HID), b, acc1);
  }
  const float bv = bro1[ncol];
  float cs = 0.0f;
#pragma unroll
  for (int e = 0; e < 8; ++e) cs += tanh_f(acc0[e] + bv) + tanh_f(acc1[e] + bv);
  float other = __shfl_down(cs, 16);
  if (lane < 16) hpart[bid * HID + ncol] = cs + other;
}

__global__ __launch_bounds__(128) void readout2_kernel(
    const float* __restrict__ tacc, const float* __restrict__ Wro2,
    const float* __restrict__ bro2, float* __restrict__ out) {
  __shared__ float tl[HID];
  const int b = blockIdx.x, t = threadIdx.x;
  tl[t] = tacc[b * HID + t];
  __syncthreads();
  float acc = (float)NVERT * bro2[t];
#pragma unroll 8
  for (int k = 0; k < HID; ++k) acc += tl[k] * Wro2[k * HID + t];
  out[b * HID + t] = acc;
}

// ------------------------------------------------------------------ host -----
extern "C" void kernel_launch(void* const* d_in, const int* in_sizes, int n_in,
                              void* d_out, int out_size, void* d_ws, size_t ws_size,
                              hipStream_t stream) {
  (void)in_sizes; (void)n_in; (void)out_size; (void)ws_size;
  const float* jets  = (const float*)d_in[0];
  const float* W_emb = (const float*)d_in[1];
  const float* b_emb = (const float*)d_in[2];
  const float* Wm    = (const float*)d_in[3];
  const float* bm    = (const float*)d_in[4];
  const float* Wz    = (const float*)d_in[5];
  const float* bz    = (const float*)d_in[6];
  const float* Wr    = (const float*)d_in[7];
  const float* br    = (const float*)d_in[8];
  const float* Wh    = (const float*)d_in[9];
  const float* bh    = (const float*)d_in[10];
  const float* Wro1  = (const float*)d_in[11];
  const float* bro1  = (const float*)d_in[12];
  const float* Wro2  = (const float*)d_in[13];
  const float* bro2  = (const float*)d_in[14];
  float* out = (float*)d_out;

  char* p = (char*)d_ws;
  auto carve = [&](size_t bytes) {
    char* q = p;
    p += (bytes + 255) & ~(size_t)255;
    return (void*)q;
  };
  half_t* hA    = (half_t*)carve((size_t)ROWS * HID * sizeof(half_t));
  half_t* hB    = (half_t*)carve((size_t)ROWS * HID * sizeof(half_t));
  float*  hsum0 = (float*) carve((size_t)BATCH * HID * sizeof(float));
  float*  hsum1 = (float*) carve((size_t)BATCH * HID * sizeof(float));
  float*  czb   = (float*) carve((size_t)BATCH * HID * sizeof(float));
  float*  crb   = (float*) carve((size_t)BATCH * HID * sizeof(float));
  float*  chb   = (float*) carve((size_t)BATCH * HID * sizeof(float));
  float*  taccb = (float*) carve((size_t)BATCH * HID * sizeof(float));
  float*  hpart = (float*) carve((size_t)GBLK * HID * sizeof(float));
  half_t* WzBig[NITER]; half_t* WrBig[NITER]; half_t* WhBig[NITER];
  half_t* WmP[NITER];   half_t* WzMid[NITER]; half_t* WrMid[NITER]; half_t* WhMid[NITER];
  for (int i = 0; i < NITER; ++i) {
    WzBig[i] = (half_t*)carve(BIGSZ * sizeof(half_t));
    WrBig[i] = (half_t*)carve(BIGSZ * sizeof(half_t));
    WhBig[i] = (half_t*)carve(BIGSZ * sizeof(half_t));
    WmP[i]   = (half_t*)carve(MIDSZ * sizeof(half_t));
    WzMid[i] = (half_t*)carve(MIDSZ * sizeof(half_t));
    WrMid[i] = (half_t*)carve(MIDSZ * sizeof(half_t));
    WhMid[i] = (half_t*)carve(MIDSZ * sizeof(half_t));
  }
  half_t* Wro1P = (half_t*)carve(MIDSZ * sizeof(half_t));

  // pack weights into WMMA B-fragment order (f16)
  for (int i = 0; i < NITER; ++i) {
    const float* wzi = Wz + (size_t)i * GINUM * HID;
    const float* wri = Wr + (size_t)i * GINUM * HID;
    const float* whi = Wh + (size_t)i * GINUM * HID;
    pack_kernel<<<BIGSZ / 256, 256, 0, stream>>>(WzBig[i], wzi, KC_BIG, 1);
    pack_kernel<<<BIGSZ / 256, 256, 0, stream>>>(WrBig[i], wri, KC_BIG, 1);
    pack_kernel<<<BIGSZ / 256, 256, 0, stream>>>(WhBig[i], whi, KC_BIG, 1);
    pack_kernel<<<MIDSZ / 256, 256, 0, stream>>>(WmP[i], Wm + (size_t)i * HID * HID, KC_H, 0);
    pack_kernel<<<MIDSZ / 256, 256, 0, stream>>>(WzMid[i], wzi, KC_H, 2);
    pack_kernel<<<MIDSZ / 256, 256, 0, stream>>>(WrMid[i], wri, KC_H, 2);
    pack_kernel<<<MIDSZ / 256, 256, 0, stream>>>(WhMid[i], whi, KC_H, 2);
  }
  pack_kernel<<<MIDSZ / 256, 256, 0, stream>>>(Wro1P, Wro1, KC_H, 0);

  // embedding + initial vertex-sum
  embed_kernel<<<GBLK, 256, 0, stream>>>(jets, W_emb, b_emb, hA, hpart);
  reduce8_kernel<<<BATCH, 128, 0, stream>>>(hsum0, hpart);

  half_t* hcur = hA; half_t* hnxt = hB;
  float* hs[2] = {hsum0, hsum1};
  int cur = 0;
  for (int i = 0; i < NITER; ++i) {
    message_kernel<<<BATCH / 16, 256, 0, stream>>>(
        hs[cur], WmP[i], bm + (size_t)i * HID,
        WzMid[i], WrMid[i], WhMid[i],
        bz + (size_t)i * HID, br + (size_t)i * HID, bh + (size_t)i * HID,
        czb, crb, chb);
    gru_kernel<<<GBLK, 256, 0, stream>>>(
        hcur, hnxt, jets, WzBig[i], WrBig[i], WhBig[i], czb, crb, chb, hpart);
    if (i + 1 < NITER)
      reduce8_kernel<<<BATCH, 128, 0, stream>>>(hs[cur ^ 1], hpart);
    half_t* t = hcur; hcur = hnxt; hnxt = t;
    cur ^= 1;
  }

  readout1_kernel<<<GBLK, 256, 0, stream>>>(hcur, Wro1P, bro1, hpart);
  reduce8_kernel<<<BATCH, 128, 0, stream>>>(taccb, hpart);
  readout2_kernel<<<BATCH, 128, 0, stream>>>(taccb, Wro2, bro2, out);
}
